// MultiHeadSelfAttention_25821343383622
// MI455X (gfx1250) — matrix operand order repaired and checked by reading, not compiled, not run on hardware
//
#include <hip/hip_runtime.h>
#include <hip/hip_bf16.h>

typedef __attribute__((ext_vector_type(16))) _Float16 v16h;
typedef __attribute__((ext_vector_type(8)))  _Float16 v8h;
typedef __attribute__((ext_vector_type(8)))  float    v8f;

#define EMBED 1024
#define HEADS 16
#define HDIM  64
#define BATCH 4
#define SEQ   2048
#define MROWS (BATCH * SEQ)   // 8192

__device__ __forceinline__ v8f wmma_f16(v16h a, v16h b, v8f c) {
    return __builtin_amdgcn_wmma_f32_16x16x32_f16(
        false, a, false, b, (short)0, c, false, false);
}

// ---------------------------------------------------------------------------
// f32 -> f16 conversion
// ---------------------------------------------------------------------------
__global__ void cvt_f32_to_f16(const float* __restrict__ in,
                               _Float16* __restrict__ out, int n) {
    int i = blockIdx.x * blockDim.x + threadIdx.x;
    int stride = gridDim.x * blockDim.x;
    for (; i < n; i += stride) out[i] = (_Float16)in[i];
}

// ---------------------------------------------------------------------------
// Hierarchical NT GEMM: C[m,n] = sum_k A[m,k]*Bw[n,k] + bias[n]
//   Block tile 128(M) x 128(N), double-buffered LDS staging (8KB+8KB per buf).
//   8 waves in a 2(M) x 4(N) grid; each wave: 64x32 register tile = 4x2 WMMA
//   accumulators, 8 WMMAs per 32-wide k-step, 4x/2x fragment reuse from LDS.
// Store modes:
//   0: f16 scatter to [B,H,T,64]   (Q,K)
//   1: f16 scatter to [B,H,64,T]   (V transposed)
//   3: f32 plain [M,N] + bias      (output projection)
// ---------------------------------------------------------------------------
__global__ __launch_bounds__(256) void gemm_nt(
    const _Float16* __restrict__ A, const _Float16* __restrict__ Bw,
    const float* __restrict__ bias, int M, int N, int K, int mode,
    _Float16* __restrict__ out16, float* __restrict__ out32)
{
    __shared__ _Float16 sA[2][128 * 32];   // [row m][k]  8KB per buffer
    __shared__ _Float16 sB[2][128 * 32];   // [col n][k]  8KB per buffer

    const int lane = threadIdx.x;          // 0..31
    const int wave = threadIdx.y;          // 0..7
    const int tid  = wave * 32 + lane;
    const int row  = lane & 15;
    const int hi   = lane >> 4;

    const int tilesN = N >> 7;                    // 128-wide N tiles
    const int tmBase = (blockIdx.x / tilesN) * 128;
    const int nbBase = (blockIdx.x % tilesN) * 128;

    // cooperative stage: each thread owns 16 halves (2x b128) of A and of B
    const int ar = tid >> 1;                      // 0..127 (tile row / col)
    const int ak = (tid & 1) * 16;                // 0 or 16
    const _Float16* Ag = A  + (size_t)(tmBase + ar) * K + ak;
    const _Float16* Bg = Bw + (size_t)(nbBase + ar) * K + ak;

    const int wm = wave & 1;                      // M wave coord (0..1)
    const int wn = wave >> 1;                     // N wave coord (0..3)

    // preload k-tile 0
    v8h ra0 = *(const v8h*)(Ag);
    v8h ra1 = *(const v8h*)(Ag + 8);
    v8h rb0 = *(const v8h*)(Bg);
    v8h rb1 = *(const v8h*)(Bg + 8);
    *(v8h*)&sA[0][ar * 32 + ak]     = ra0;
    *(v8h*)&sA[0][ar * 32 + ak + 8] = ra1;
    *(v8h*)&sB[0][ar * 32 + ak]     = rb0;
    *(v8h*)&sB[0][ar * 32 + ak + 8] = rb1;
    __syncthreads();

    v8f c[4][2];
    #pragma unroll
    for (int i = 0; i < 4; ++i)
        #pragma unroll
        for (int j = 0; j < 2; ++j) c[i][j] = (v8f){};

    int buf = 0;
    for (int kb = 0; kb < K; kb += 32, buf ^= 1) {
        const bool more = (kb + 32) < K;
        if (more) {                                   // fetch next k-tile
            ra0 = *(const v8h*)(Ag + kb + 32);
            ra1 = *(const v8h*)(Ag + kb + 40);
            rb0 = *(const v8h*)(Bg + kb + 32);
            rb1 = *(const v8h*)(Bg + kb + 40);
            if (kb + 64 < K) {                        // L2 prefetch 2 tiles out
                __builtin_prefetch(Ag + kb + 64, 0, 0);
                __builtin_prefetch(Bg + kb + 64, 0, 0);
            }
        }

        // compute from current LDS buffer: 4 A frags, 2 B frags, 8 WMMAs
        v16h a[4], b[2];
        #pragma unroll
        for (int i = 0; i < 4; ++i) {
            const int r = (64 * wm + 16 * i + row) * 32;
            *(v8h*)&a[i]       = *(const v8h*)&sA[buf][r + 8 * hi];
            *((v8h*)&a[i] + 1) = *(const v8h*)&sA[buf][r + 16 + 8 * hi];
        }
        #pragma unroll
        for (int j = 0; j < 2; ++j) {
            const int n = (32 * wn + 16 * j + row) * 32;
            *(v8h*)&b[j]       = *(const v8h*)&sB[buf][n + 8 * hi];
            *((v8h*)&b[j] + 1) = *(const v8h*)&sB[buf][n + 16 + 8 * hi];
        }
        #pragma unroll
        for (int i = 0; i < 4; ++i)
            #pragma unroll
            for (int j = 0; j < 2; ++j)
                c[i][j] = wmma_f16(a[i], b[j], c[i][j]);

        if (more) {                                   // stage next into LDS
            *(v8h*)&sA[buf ^ 1][ar * 32 + ak]     = ra0;
            *(v8h*)&sA[buf ^ 1][ar * 32 + ak + 8] = ra1;
            *(v8h*)&sB[buf ^ 1][ar * 32 + ak]     = rb0;
            *(v8h*)&sB[buf ^ 1][ar * 32 + ak + 8] = rb1;
        }
        __syncthreads();
    }

    // store with bias, per mode
    #pragma unroll
    for (int j = 0; j < 2; ++j) {
        const int nCol = nbBase + 32 * wn + 16 * j + row;
        const float bv = bias ? bias[nCol] : 0.0f;
        #pragma unroll
        for (int i = 0; i < 4; ++i) {
            #pragma unroll
            for (int r = 0; r < 8; ++r) {
                const int m = tmBase + 64 * wm + 16 * i + r + 8 * hi;
                const float val = c[i][j][r] + bv;
                if (mode == 3) {
                    out32[(size_t)m * N + nCol] = val;
                } else {
                    const int bb = m >> 11;           // m / SEQ
                    const int t  = m & (SEQ - 1);
                    const int h  = nCol >> 6;
                    const int d  = nCol & (HDIM - 1);
                    if (mode == 0)
                        out16[(((size_t)(bb * HEADS + h)) * SEQ + t) * HDIM + d] = (_Float16)val;
                    else
                        out16[(((size_t)(bb * HEADS + h)) * HDIM + d) * SEQ + t] = (_Float16)val;
                }
            }
        }
    }
}

// ---------------------------------------------------------------------------
// Flash attention. Block = one (b,h) x 128 query rows; wave = 16 query rows.
// 32-key tiles of K (32x64) and V^T (64x32) are cooperatively staged in
// double-buffered LDS and shared by all 8 waves (8x less global K/V traffic).
// Per tile: S = Q K^T (4 WMMA over the two 16-key halves), online softmax
// (16-lane xor reductions), P transposed C->A layout via per-wave LDS
// (in-order LDS pipe + s_wait_dscnt), O += P V (4 WMMA).
// ---------------------------------------------------------------------------
__global__ __launch_bounds__(256) void attention(
    const _Float16* __restrict__ Q, const _Float16* __restrict__ Kt,
    const _Float16* __restrict__ Vt, _Float16* __restrict__ ctx)
{
    __shared__ _Float16 sK[2][32 * 64];     // [key][d]   4KB per buffer
    __shared__ _Float16 sV[2][64 * 32];     // [d][key]   4KB per buffer
    __shared__ _Float16 sP[8][16 * 32];     // per-wave P transpose scratch

    const int lane = threadIdx.x;
    const int wave = threadIdx.y;
    const int tid  = wave * 32 + lane;
    const int row  = lane & 15;
    const int hi   = lane >> 4;

    const int bh = blockIdx.x >> 4;          // 0..63  (b*16+h)
    const int qb = blockIdx.x & 15;          // 128-row query block
    const int qt = qb * 8 + wave;            // this wave's 16-row q tile

    const _Float16* Qp = Q  + (size_t)bh * SEQ * HDIM;
    const _Float16* Kp = Kt + (size_t)bh * SEQ * HDIM;
    const _Float16* Vp = Vt + (size_t)bh * HDIM * SEQ;

    // cooperative K/V staging: each thread owns 8 halves of each tile
    const int idx8 = tid * 8;
    const int kr = idx8 >> 6, kk = idx8 & 63;   // K tile coords [32][64]
    const int dr = idx8 >> 5, dk = idx8 & 31;   // V tile coords [64][32]

    // Q A-fragments, pre-scaled by 1/sqrt(64)
    v16h a0, a1;
    const _Float16* Qr = Qp + (size_t)(qt * 16 + row) * HDIM;
    *(v8h*)&a0       = *(const v8h*)(Qr + 8 * hi);
    *((v8h*)&a0 + 1) = *(const v8h*)(Qr + 16 + 8 * hi);
    *(v8h*)&a1       = *(const v8h*)(Qr + 32 + 8 * hi);
    *((v8h*)&a1 + 1) = *(const v8h*)(Qr + 48 + 8 * hi);
    #pragma unroll
    for (int i = 0; i < 16; ++i) {
        a0[i] *= (_Float16)0.125f;
        a1[i] *= (_Float16)0.125f;
    }

    v8f o[4] = {{}, {}, {}, {}};
    float mr[8], lr[8];
    #pragma unroll
    for (int r = 0; r < 8; ++r) { mr[r] = -1e30f; lr[r] = 0.0f; }

    _Float16* myP = &sP[wave][0];

    // preload key tile 0
    v8h rK = *(const v8h*)(Kp + (size_t)kr * HDIM + kk);
    v8h rV = *(const v8h*)(Vp + (size_t)dr * SEQ + dk);
    *(v8h*)&sK[0][kr * 64 + kk] = rK;
    *(v8h*)&sV[0][dr * 32 + dk] = rV;
    __syncthreads();

    int buf = 0;
    for (int kt = 0; kt < SEQ; kt += 32, buf ^= 1) {
        const bool more = (kt + 32) < SEQ;
        if (more) {
            rK = *(const v8h*)(Kp + (size_t)(kt + 32 + kr) * HDIM + kk);
            rV = *(const v8h*)(Vp + (size_t)dr * SEQ + kt + 32 + dk);
        }

        // ---- S = (Q/8) K^T: keys [kt,kt+16) -> s0, [kt+16,kt+32) -> s1 ----
        v8f s0 = {}, s1 = {};
        {
            v16h b;
            const int k0 = row * 64;            // key = kt + col
            const int k1 = (16 + row) * 64;     // key = kt + 16 + col
            *(v8h*)&b       = *(const v8h*)&sK[buf][k0 + 8 * hi];
            *((v8h*)&b + 1) = *(const v8h*)&sK[buf][k0 + 16 + 8 * hi];
            s0 = wmma_f16(a0, b, s0);
            *(v8h*)&b       = *(const v8h*)&sK[buf][k0 + 32 + 8 * hi];
            *((v8h*)&b + 1) = *(const v8h*)&sK[buf][k0 + 48 + 8 * hi];
            s0 = wmma_f16(a1, b, s0);
            *(v8h*)&b       = *(const v8h*)&sK[buf][k1 + 8 * hi];
            *((v8h*)&b + 1) = *(const v8h*)&sK[buf][k1 + 16 + 8 * hi];
            s1 = wmma_f16(a0, b, s1);
            *(v8h*)&b       = *(const v8h*)&sK[buf][k1 + 32 + 8 * hi];
            *((v8h*)&b + 1) = *(const v8h*)&sK[buf][k1 + 48 + 8 * hi];
            s1 = wmma_f16(a1, b, s1);
        }

        // ---- online softmax (rows live across a 16-lane half) ----
        float sc[8];
        #pragma unroll
        for (int r = 0; r < 8; ++r) {
            float rm = fmaxf(s0[r], s1[r]);
            #pragma unroll
            for (int msk = 1; msk < 16; msk <<= 1)
                rm = fmaxf(rm, __shfl_xor(rm, msk, 32));
            const float mn = fmaxf(mr[r], rm);
            const float p0 = __expf(s0[r] - mn);
            const float p1 = __expf(s1[r] - mn);
            float rs = p0 + p1;
            #pragma unroll
            for (int msk = 1; msk < 16; msk <<= 1)
                rs += __shfl_xor(rs, msk, 32);
            sc[r] = __expf(mr[r] - mn);
            lr[r] = lr[r] * sc[r] + rs;
            mr[r] = mn;
            s0[r] = p0;
            s1[r] = p1;
        }
        #pragma unroll
        for (int ch = 0; ch < 4; ++ch)
            #pragma unroll
            for (int r = 0; r < 8; ++r) o[ch][r] *= sc[r];

        // ---- P: C layout -> A layout via per-wave LDS scratch ----
        #pragma unroll
        for (int r = 0; r < 8; ++r) {
            myP[(r + 8 * hi) * 32 + row]      = (_Float16)s0[r];
            myP[(r + 8 * hi) * 32 + 16 + row] = (_Float16)s1[r];
        }
        __builtin_amdgcn_wave_barrier();
        asm volatile("s_wait_dscnt 0" ::: "memory");
        v16h p;
        *(v8h*)&p       = *(const v8h*)(myP + row * 32 + 8 * hi);
        *((v8h*)&p + 1) = *(const v8h*)(myP + row * 32 + 16 + 8 * hi);
        __builtin_amdgcn_wave_barrier();

        // ---- O += P(16x32) @ V(32x64): 4 WMMAs over d-chunks ----
        #pragma unroll
        for (int ch = 0; ch < 4; ++ch) {
            v16h b;
            const int d = (ch * 16 + row) * 32;
            *(v8h*)&b       = *(const v8h*)&sV[buf][d + 8 * hi];
            *((v8h*)&b + 1) = *(const v8h*)&sV[buf][d + 16 + 8 * hi];
            o[ch] = wmma_f16(p, b, o[ch]);
        }

        if (more) {                         // stage next tile
            *(v8h*)&sK[buf ^ 1][kr * 64 + kk] = rK;
            *(v8h*)&sV[buf ^ 1][dr * 32 + dk] = rV;
        }
        __syncthreads();
    }

    // ---- normalize and store context as [B*T, 1024] f16 ----
    const int bb = bh >> 4, h = bh & 15;
    #pragma unroll
    for (int ch = 0; ch < 4; ++ch) {
        #pragma unroll
        for (int r = 0; r < 8; ++r) {
            const int t = qt * 16 + r + 8 * hi;
            const int d = ch * 16 + row;
            const float val = o[ch][r] / lr[r];
            ctx[((size_t)(bb * SEQ + t)) * EMBED + h * HDIM + d] = (_Float16)val;
        }
    }
}

// ---------------------------------------------------------------------------
// Host-side orchestration
// ---------------------------------------------------------------------------
extern "C" void kernel_launch(void* const* d_in, const int* in_sizes, int n_in,
                              void* d_out, int out_size, void* d_ws, size_t ws_size,
                              hipStream_t stream) {
    const float* x   = (const float*)d_in[0];
    const float* W_q = (const float*)d_in[1];
    const float* b_q = (const float*)d_in[2];
    const float* W_k = (const float*)d_in[3];
    const float* b_k = (const float*)d_in[4];
    const float* W_v = (const float*)d_in[5];
    const float* b_v = (const float*)d_in[6];
    const float* W_o = (const float*)d_in[7];
    const float* b_o = (const float*)d_in[8];
    float* out = (float*)d_out;

    char* ws = (char*)d_ws;
    const size_t XH  = (size_t)MROWS * EMBED * 2;                 // 16 MB
    const size_t WH  = (size_t)EMBED * EMBED * 2;                 //  2 MB
    const size_t QKV = (size_t)BATCH * HEADS * SEQ * HDIM * 2;    // 16 MB
    _Float16* xh  = (_Float16*)(ws);
    _Float16* ctx = (_Float16*)(ws);                              // alias xh
    _Float16* Wqh = (_Float16*)(ws + XH);
    _Float16* Wkh = (_Float16*)(ws + XH + WH);
    _Float16* Wvh = (_Float16*)(ws + XH + 2 * WH);
    _Float16* Woh = (_Float16*)(ws + XH + 3 * WH);
    _Float16* Qh  = (_Float16*)(ws + XH + 4 * WH);
    _Float16* Kh  = (_Float16*)(ws + XH + 4 * WH + QKV);
    _Float16* Vth = (_Float16*)(ws + XH + 4 * WH + 2 * QKV);

    cvt_f32_to_f16<<<2048, 256, 0, stream>>>(x, xh, MROWS * EMBED);
    cvt_f32_to_f16<<<512, 256, 0, stream>>>(W_q, Wqh, EMBED * EMBED);
    cvt_f32_to_f16<<<512, 256, 0, stream>>>(W_k, Wkh, EMBED * EMBED);
    cvt_f32_to_f16<<<512, 256, 0, stream>>>(W_v, Wvh, EMBED * EMBED);
    cvt_f32_to_f16<<<512, 256, 0, stream>>>(W_o, Woh, EMBED * EMBED);

    dim3 blk(32, 8);
    const int gemmBlocks = (MROWS / 128) * (EMBED / 128);   // 64*8 = 512
    gemm_nt<<<gemmBlocks, blk, 0, stream>>>(xh, Wqh, b_q, MROWS, EMBED, EMBED, 0, Qh, nullptr);
    gemm_nt<<<gemmBlocks, blk, 0, stream>>>(xh, Wkh, b_k, MROWS, EMBED, EMBED, 0, Kh, nullptr);
    gemm_nt<<<gemmBlocks, blk, 0, stream>>>(xh, Wvh, b_v, MROWS, EMBED, EMBED, 1, Vth, nullptr);

    // 64 (b,h) x 16 query blocks of 128 rows = 1024 blocks
    attention<<<1024, blk, 0, stream>>>(Qh, Kh, Vth, ctx);

    gemm_nt<<<gemmBlocks, blk, 0, stream>>>(ctx, Woh, b_o, MROWS, EMBED, EMBED, 3, nullptr, out);
}
